// SparseAttention_11622181503356
// MI455X (gfx1250) — matrix operand order repaired and checked by reading, not compiled, not run on hardware
//
#include <hip/hip_runtime.h>
#include <stdint.h>

typedef unsigned short ushort_t;
typedef __attribute__((ext_vector_type(16))) __bf16 v16bf;
typedef __attribute__((ext_vector_type(8)))  float  v8f;

#define WMMA_BF16(a, b, c) \
    __builtin_amdgcn_wmma_f32_16x16x32_bf16(false, (a), false, (b), (short)0, (c), false, false)

#define B_SZ   2
#define T_SEQ  2048
#define DMODEL 1024
#define NH     16
#define HD     64
#define WIN    16
#define GN     128   /* T / stride */
#define QB     32    /* queries per attention block (2 waves x 16) */
#define LK     48    /* local key staging: QB + WIN, padded */
#define FRAG   520   /* swizzled B fragment stride in ushorts (512 + 8 pad) */

__device__ __forceinline__ ushort_t f2b(float f) {
    union { float f; uint32_t u; } x; x.f = f;
    uint32_t u = x.u;
    uint32_t r = u + 0x7FFFu + ((u >> 16) & 1u);
    return (ushort_t)(r >> 16);
}

// ---------------------------------------------------------------------------
// WMMA operand helpers
// ---------------------------------------------------------------------------
// A-operand: 16x32 bf16, row-major source with row stride ld (elements).
// lanes 0-15: row=lane, K = {0..15}; lanes 16-31: row=lane-16, K = {16..31}
// The 16-element run is contiguous -> two 16B loads.
__device__ __forceinline__ v16bf load_A_frag(const ushort_t* __restrict__ base, int ld, int lane) {
    const int row = lane & 15;
    const int khi = (lane >> 4) * 16;
    const ushort_t* p = base + (size_t)row * ld;
    union { v16bf v; uint4 q[2]; } r;
    r.q[0] = *(const uint4*)(p + khi);
    r.q[1] = *(const uint4*)(p + khi + 8);
    return r.v;
}

// Swizzled B-operand storage: a 32(K)x16(N) fragment occupies FRAG ushorts;
// lane l's 16 elements (u[m] = B[kb+m][col], kb=(l>>4)*16, col=l&15) are
// contiguous at frag_base + l*16  -> two ds_load_b128 per fragment.
__device__ __forceinline__ int swizB_idx(int k, int n, int nblk) {
    int fi   = (k >> 5) * nblk + (n >> 4);
    int lane = (n & 15) + ((k >> 4) & 1) * 16;
    return fi * FRAG + lane * 16 + (k & 15);
}

__device__ __forceinline__ v16bf load_B_swiz(const ushort_t* __restrict__ fragBase, int lane) {
    union { v16bf v; uint4 q[2]; } r;
    const uint4* p = (const uint4*)(fragBase + lane * 16);
    r.q[0] = p[0];
    r.q[1] = p[1];
    return r.v;
}

__device__ __forceinline__ v8f zero_v8f() {
    v8f z = {0.f, 0.f, 0.f, 0.f, 0.f, 0.f, 0.f, 0.f};
    return z;
}

// ---------------------------------------------------------------------------
// fp32 -> bf16 convert
// ---------------------------------------------------------------------------
__global__ void cvt_f32_bf16_kernel(const float* __restrict__ src,
                                    ushort_t* __restrict__ dst, int n) {
    int i = blockIdx.x * blockDim.x + threadIdx.x;
    if (i < n) dst[i] = f2b(src[i]);
}

// ---------------------------------------------------------------------------
// Generic tiled WMMA GEMM: C[M,N] = A[M,K] * B[K,N]; A,B bf16 row-major.
// outBF16=1: store bf16 (no bias). outBF16=0: store fp32 + bias.
// Block: 256 threads (8 waves, 2x4), tile 64x128, BK=32. B staged swizzled.
// ---------------------------------------------------------------------------
__global__ __launch_bounds__(256)
void gemm_bf16_kernel(const ushort_t* __restrict__ A,
                      const ushort_t* __restrict__ Bm,
                      void* __restrict__ Cout,
                      const float* __restrict__ bias,
                      int M, int N, int K, int outBF16) {
    __shared__ ushort_t As[64][40];       // 64x32 + pad (A runs contiguous)
    __shared__ ushort_t BsSw[8 * FRAG];   // 8 swizzled 32x16 fragments

    const int tid  = threadIdx.x;
    const int lane = tid & 31;
    const int wave = tid >> 5;
    const int wm   = wave >> 2;   // 0..1
    const int wn   = wave & 3;    // 0..3
    const int bM   = blockIdx.y * 64;
    const int bN   = blockIdx.x * 128;

    v8f acc[2][2];
#pragma unroll
    for (int i = 0; i < 2; ++i)
#pragma unroll
        for (int j = 0; j < 2; ++j) acc[i][j] = zero_v8f();

    const int arow = tid >> 2;          // 0..63
    const int acol = (tid & 3) * 8;     // 0,8,16,24
    const int brow = tid >> 3;          // 0..31  (k within stage)
    const int bsub = tid & 7;           // n-subtile 0..7
    const int bj       = brow & 15;
    const int blaneOff = ((brow >> 4) & 1) * 16;

    for (int k0 = 0; k0 < K; k0 += 32) {
        // ---- stage A (row-major) ----
        *(uint4*)&As[arow][acol] =
            *(const uint4*)&A[(size_t)(bM + arow) * K + k0 + acol];

        // ---- stage B (swizzled to fragment order) ----
        union { uint4 q[2]; ushort_t u[16]; } bb;
        const ushort_t* bsrc = &Bm[(size_t)(k0 + brow) * N + bN + bsub * 16];
        bb.q[0] = *(const uint4*)(bsrc);
        bb.q[1] = *(const uint4*)(bsrc + 8);
        ushort_t* fb = &BsSw[bsub * FRAG];
#pragma unroll
        for (int e = 0; e < 16; ++e)
            fb[(blaneOff + e) * 16 + bj] = bb.u[e];

        if (k0 + 32 < K) {
            __builtin_prefetch(&A[(size_t)(bM + arow) * K + k0 + 32 + acol], 0, 3);
            __builtin_prefetch(&Bm[(size_t)(k0 + 32 + brow) * N + bN + bsub * 16], 0, 3);
        }
        __syncthreads();

        v16bf a0 = load_A_frag(&As[wm * 32][0],      40, lane);
        v16bf a1 = load_A_frag(&As[wm * 32 + 16][0], 40, lane);
        v16bf b0 = load_B_swiz(&BsSw[(wn * 2 + 0) * FRAG], lane);
        v16bf b1 = load_B_swiz(&BsSw[(wn * 2 + 1) * FRAG], lane);

        acc[0][0] = WMMA_BF16(a0, b0, acc[0][0]);
        acc[0][1] = WMMA_BF16(a0, b1, acc[0][1]);
        acc[1][0] = WMMA_BF16(a1, b0, acc[1][0]);
        acc[1][1] = WMMA_BF16(a1, b1, acc[1][1]);
        __syncthreads();
    }

    const int col0 = lane & 15;
    const int roff = (lane >> 4) * 8;
#pragma unroll
    for (int i = 0; i < 2; ++i)
#pragma unroll
        for (int j = 0; j < 2; ++j)
#pragma unroll
            for (int r = 0; r < 8; ++r) {
                int row = bM + wm * 32 + i * 16 + roff + r;
                int col = bN + wn * 32 + j * 16 + col0;
                float v = acc[i][j][r];
                if (outBF16) {
                    ((ushort_t*)Cout)[(size_t)row * N + col] = f2b(v);
                } else {
                    ((float*)Cout)[(size_t)row * N + col] = v + bias[col];
                }
            }
}

// ---------------------------------------------------------------------------
// Attention: local sliding-window (W=16, zero-padded like reference) softmax
// + global strided (128 keys) softmax; outputs summed, written as bf16
// in (B,T,D) layout. Block = 64 threads (2 waves x 16-query tiles).
// All K/V operands staged in swizzled WMMA-fragment order (wide ds loads).
// ---------------------------------------------------------------------------
__global__ __launch_bounds__(64)
void attn_kernel(const ushort_t* __restrict__ qkv, ushort_t* __restrict__ outb) {
    __shared__ ushort_t KgSw[16 * FRAG];       // K^T global: K=64(d) x N=128(g) -> 2x8 frags
    __shared__ ushort_t VgSw[16 * FRAG];       // V global:   K=128(g) x N=64(d) -> 4x4 frags
    __shared__ ushort_t KlSw[6 * FRAG];        // K^T local:  K=64(d) x N=48(j)  -> 2x3 frags
    __shared__ ushort_t VlSw[2 * 4 * FRAG];    // V local, per wave: K=32(j) x N=64(d) -> 4 frags
    __shared__ ushort_t Pw[2][16][GN + 8];     // per-wave probability staging (A-operand, row-major)

    const int tid  = threadIdx.x;
    const int lane = tid & 31;
    const int w    = tid >> 5;
    const int nblkT = T_SEQ / QB;
    const int bh = blockIdx.x / nblkT;
    const int tb = (blockIdx.x % nblkT) * QB;
    const int b  = bh / NH;
    const int h  = bh % NH;
    const float scale = 0.125f;  // 1/sqrt(64)

    const int rowStride = 3 * DMODEL;
    const ushort_t* qbase = qkv + (size_t)b * T_SEQ * rowStride + h * HD;
    const ushort_t* kbase = qbase + DMODEL;
    const ushort_t* vbase = qbase + 2 * DMODEL;

    // Stage global K^T and V (keys 0,16,...,2032), swizzled
    for (int i = tid; i < GN * HD; i += 64) {
        int g = i >> 6, d = i & 63;
        KgSw[swizB_idx(d, g, 8)] = kbase[(size_t)(g * 16) * rowStride + d];
        VgSw[swizB_idx(g, d, 4)] = vbase[(size_t)(g * 16) * rowStride + d];
    }
    // Stage local window K^T: keys tb-15 .. tb+32 (zero pad out-of-range)
    for (int i = tid; i < LK * HD; i += 64) {
        int j = i >> 6, d = i & 63;
        int kidx = tb - (WIN - 1) + j;
        ushort_t kv = 0;
        if (kidx >= 0 && kidx < T_SEQ) kv = kbase[(size_t)kidx * rowStride + d];
        KlSw[swizB_idx(d, j, 3)] = kv;
    }
    // Stage local window V per wave (wave w2 uses keys tb-15+16*w2 + 0..31)
    for (int i = tid; i < 2 * 32 * HD; i += 64) {
        int w2  = i >> 11;
        int rem = i & 2047;
        int j = rem >> 6, d = rem & 63;
        int kidx = tb - (WIN - 1) + 16 * w2 + j;
        ushort_t vv = 0;
        if (kidx >= 0 && kidx < T_SEQ) vv = vbase[(size_t)kidx * rowStride + d];
        VlSw[w2 * 4 * FRAG + swizB_idx(j, d, 4)] = vv;
    }
    __syncthreads();

    const int t0 = tb + 16 * w;
    v16bf aq[2];
    aq[0] = load_A_frag(qbase + (size_t)t0 * rowStride,      rowStride, lane);
    aq[1] = load_A_frag(qbase + (size_t)t0 * rowStride + 32, rowStride, lane);

    const int col0 = lane & 15;
    const int roff = (lane >> 4) * 8;

    // ---------------- Local attention ----------------
    v8f sl[2] = {zero_v8f(), zero_v8f()};
#pragma unroll
    for (int c = 0; c < 2; ++c) {
        v16bf bk0 = load_B_swiz(&KlSw[(c * 3 + w) * FRAG],     lane);
        v16bf bk1 = load_B_swiz(&KlSw[(c * 3 + w + 1) * FRAG], lane);
        sl[0] = WMMA_BF16(aq[c], bk0, sl[0]);
        sl[1] = WMMA_BF16(aq[c], bk1, sl[1]);
    }
    // scale + window mask (keys jj in [m, m+15]; zero-padded keys keep score 0)
#pragma unroll
    for (int r = 0; r < 8; ++r) {
        int m = r + roff;
#pragma unroll
        for (int f = 0; f < 2; ++f) {
            int jj = f * 16 + col0;
            float s = sl[f][r] * scale;
            if (jj < m || jj > m + 15) s = -3.0e38f;
            sl[f][r] = s;
        }
    }
    // row softmax (row = 16 lanes in half-wave x 2 fragments)
#pragma unroll
    for (int r = 0; r < 8; ++r) {
        float mx = fmaxf(sl[0][r], sl[1][r]);
        for (int off = 1; off < 16; off <<= 1) mx = fmaxf(mx, __shfl_xor(mx, off, 32));
        float e0 = __expf(sl[0][r] - mx);
        float e1 = __expf(sl[1][r] - mx);
        float sm = e0 + e1;
        for (int off = 1; off < 16; off <<= 1) sm += __shfl_xor(sm, off, 32);
        float inv = 1.0f / sm;
        sl[0][r] = e0 * inv;
        sl[1][r] = e1 * inv;
    }
    // bounce P to LDS, reload as A fragment
#pragma unroll
    for (int r = 0; r < 8; ++r) {
        int m = r + roff;
        Pw[w][m][col0]      = f2b(sl[0][r]);
        Pw[w][m][16 + col0] = f2b(sl[1][r]);
    }
    asm volatile("s_wait_dscnt 0x0" ::: "memory");

    v8f acc[4];
#pragma unroll
    for (int n = 0; n < 4; ++n) acc[n] = zero_v8f();
    {
        v16bf ap = load_A_frag(&Pw[w][0][0], GN + 8, lane);
#pragma unroll
        for (int n = 0; n < 4; ++n) {
            v16bf bv = load_B_swiz(&VlSw[(w * 4 + n) * FRAG], lane);
            acc[n] = WMMA_BF16(ap, bv, acc[n]);
        }
    }

    // ---------------- Global attention ----------------
    v8f sg[8];
#pragma unroll
    for (int g = 0; g < 8; ++g) sg[g] = zero_v8f();
#pragma unroll
    for (int c = 0; c < 2; ++c)
#pragma unroll
        for (int g = 0; g < 8; ++g) {
            v16bf bk = load_B_swiz(&KgSw[(c * 8 + g) * FRAG], lane);
            sg[g] = WMMA_BF16(aq[c], bk, sg[g]);
        }
#pragma unroll
    for (int r = 0; r < 8; ++r) {
        float mx = -3.0e38f;
#pragma unroll
        for (int g = 0; g < 8; ++g) {
            sg[g][r] *= scale;
            mx = fmaxf(mx, sg[g][r]);
        }
        for (int off = 1; off < 16; off <<= 1) mx = fmaxf(mx, __shfl_xor(mx, off, 32));
        float sm = 0.f;
#pragma unroll
        for (int g = 0; g < 8; ++g) {
            float e = __expf(sg[g][r] - mx);
            sg[g][r] = e;
            sm += e;
        }
        for (int off = 1; off < 16; off <<= 1) sm += __shfl_xor(sm, off, 32);
        float inv = 1.0f / sm;
#pragma unroll
        for (int g = 0; g < 8; ++g) sg[g][r] *= inv;
    }
#pragma unroll
    for (int r = 0; r < 8; ++r) {
        int m = r + roff;
#pragma unroll
        for (int g = 0; g < 8; ++g) Pw[w][m][g * 16 + col0] = f2b(sg[g][r]);
    }
    asm volatile("s_wait_dscnt 0x0" ::: "memory");

#pragma unroll
    for (int c = 0; c < 4; ++c) {
        v16bf ap = load_A_frag(&Pw[w][0][32 * c], GN + 8, lane);
#pragma unroll
        for (int n = 0; n < 4; ++n) {
            v16bf bv = load_B_swiz(&VgSw[(c * 4 + n) * FRAG], lane);
            acc[n] = WMMA_BF16(ap, bv, acc[n]);
        }
    }

    // Write out (B,T,D) bf16
#pragma unroll
    for (int n = 0; n < 4; ++n)
#pragma unroll
        for (int r = 0; r < 8; ++r) {
            int t  = t0 + r + roff;
            int cc = h * HD + n * 16 + col0;
            outb[(size_t)(b * T_SEQ + t) * DMODEL + cc] = f2b(acc[n][r]);
        }
}

// ---------------------------------------------------------------------------
extern "C" void kernel_launch(void* const* d_in, const int* in_sizes, int n_in,
                              void* d_out, int out_size, void* d_ws, size_t ws_size,
                              hipStream_t stream) {
    const float* x      = (const float*)d_in[0];
    const float* w_qkv  = (const float*)d_in[1];
    const float* w_proj = (const float*)d_in[2];
    const float* b_proj = (const float*)d_in[3];
    // window=16, stride=16 per setup_inputs (compile-time tiled)

    const size_t nX    = (size_t)B_SZ * T_SEQ * DMODEL;          // 4,194,304
    const size_t nWqkv = (size_t)DMODEL * 3 * DMODEL;            // 3,145,728
    const size_t nWprj = (size_t)DMODEL * DMODEL;                // 1,048,576
    const size_t nQkv  = (size_t)B_SZ * T_SEQ * 3 * DMODEL;      // 12,582,912

    ushort_t* ws = (ushort_t*)d_ws;
    ushort_t* xb     = ws;
    ushort_t* wqkvb  = xb + nX;
    ushort_t* wprojb = wqkvb + nWqkv;
    ushort_t* qkvb   = wprojb + nWprj;
    ushort_t* attnb  = qkvb + nQkv;

    // 1) fp32 -> bf16 conversions
    cvt_f32_bf16_kernel<<<(int)((nX + 255) / 256), 256, 0, stream>>>(x, xb, (int)nX);
    cvt_f32_bf16_kernel<<<(int)((nWqkv + 255) / 256), 256, 0, stream>>>(w_qkv, wqkvb, (int)nWqkv);
    cvt_f32_bf16_kernel<<<(int)((nWprj + 255) / 256), 256, 0, stream>>>(w_proj, wprojb, (int)nWprj);

    // 2) QKV GEMM: (4096 x 1024) * (1024 x 3072) -> bf16
    {
        dim3 grid(3 * DMODEL / 128, (B_SZ * T_SEQ) / 64);
        gemm_bf16_kernel<<<grid, 256, 0, stream>>>(xb, wqkvb, qkvb, nullptr,
                                                   B_SZ * T_SEQ, 3 * DMODEL, DMODEL, 1);
    }

    // 3) Attention (local window + strided global)
    {
        int grid = B_SZ * NH * (T_SEQ / QB);  // 2048 blocks of 64 threads
        attn_kernel<<<grid, 64, 0, stream>>>(qkvb, attnb);
    }

    // 4) Output projection: (4096 x 1024) * (1024 x 1024) + bias -> fp32 d_out
    {
        dim3 grid(DMODEL / 128, (B_SZ * T_SEQ) / 64);
        gemm_bf16_kernel<<<grid, 256, 0, stream>>>(attnb, wprojb, d_out, b_proj,
                                                   B_SZ * T_SEQ, DMODEL, DMODEL, 0);
    }
}
